// Concatenation_90701119357422
// MI455X (gfx1250) — matrix operand order repaired and checked by reading, not compiled, not run on hardware
//
#include <hip/hip_runtime.h>
#include <hip/hip_bf16.h>

// Problem constants (fixed by the reference)
#define NTOT 262144
#define BSZ  64
#define KRET 16
#define HDIM 256
#define RDIM 512

typedef __attribute__((ext_vector_type(16))) __bf16 v16bf;
typedef __attribute__((ext_vector_type(8)))  float  v8f;

// Split f32 into bf16 hi (truncated, exact remainder) + bf16 lo (RNE of remainder).
__device__ __forceinline__ void split2(float x, __bf16& hi, __bf16& lo) {
    unsigned u = __builtin_bit_cast(unsigned, x);
    float hf = __builtin_bit_cast(float, u & 0xffff0000u);
    hi = __builtin_bit_cast(__bf16, (unsigned short)(u >> 16));
    lo = (__bf16)(x - hf);   // exact residual, rounded to bf16
}

__device__ __forceinline__ void cvt4(float4 xv, int o, v16bf& ahi, v16bf& alo) {
    __bf16 hi, lo;
    split2(xv.x, hi, lo); ahi[o+0] = hi; alo[o+0] = lo;
    split2(xv.y, hi, lo); ahi[o+1] = hi; alo[o+1] = lo;
    split2(xv.z, hi, lo); ahi[o+2] = hi; alo[o+2] = lo;
    split2(xv.w, hi, lo); ahi[o+3] = hi; alo[o+3] = lo;
}

// ---------------------------------------------------------------------------
// Stage 1: P[b][n] = bl[n] + Wl2[n,:] . ( bp + Wp . mean_k ret_feat[b,k,:] )
// 64 blocks x 256 threads, full f32. ~25 MFLOP total (noise).
// ---------------------------------------------------------------------------
__global__ __launch_bounds__(256) void pooled_kernel(
    const float* __restrict__ ret_feat, const float* __restrict__ Wp,
    const float* __restrict__ bp, const float* __restrict__ Wl,
    const float* __restrict__ bl, float* __restrict__ P) {
    __shared__ float smean[RDIM];
    __shared__ float sretH[HDIM];
    const int b = blockIdx.x, tid = threadIdx.x;

    for (int r = tid; r < RDIM; r += 256) {
        float s = 0.f;
        #pragma unroll
        for (int j = 0; j < KRET; ++j)
            s += ret_feat[((size_t)b * KRET + j) * RDIM + r];
        smean[r] = s * (1.0f / KRET);
    }
    __syncthreads();
    {   // retH[n] = bp[n] + Wp[n,:] . smean     (tid == n, HDIM == 256)
        float s = bp[tid];
        const float* w = Wp + (size_t)tid * RDIM;
        for (int r = 0; r < RDIM; ++r) s += w[r] * smean[r];
        sretH[tid] = s;
    }
    __syncthreads();
    {   // P[b][n] = bl[n] + Wl[n, H: ] . retH
        float s = bl[tid];
        const float* w = Wl + (size_t)tid * (2 * HDIM) + HDIM;
        for (int k = 0; k < HDIM; ++k) s += w[k] * sretH[k];
        P[(size_t)b * HDIM + tid] = s;
    }
}

// ---------------------------------------------------------------------------
// Stage 2: pack Wl1 (= Wl[:, :H]) into bf16 hi/lo in the WMMA B-operand
// per-lane layout, in this code's contiguous k order (A uses the same).  B is 32x16 (KxN): lane l holds col N = l&15,
// K = (l>>4)*16 + e  (e = 0..15).  Stored as
//   Bx[(((hh*8 + t)*8 + c)*32 + l)*16 + e],  n = hh*128 + t*16 + (l&15),
//   k = c*32 + (l>>4)*16 + e.
// B[k][n] = Wl1.T[k][n] = Wl[n][k].
// ---------------------------------------------------------------------------
__global__ __launch_bounds__(256) void pack_kernel(
    const float* __restrict__ Wl,
    unsigned short* __restrict__ Bhi, unsigned short* __restrict__ Blo) {
    const int w = blockIdx.x * 256 + threadIdx.x;   // 0..4095 = (hh,t,c,l)
    const int l  = w & 31;
    const int c  = (w >> 5) & 7;
    const int t  = (w >> 8) & 7;
    const int hh = w >> 11;
    const int n  = hh * 128 + t * 16 + (l & 15);
    const int kb = c * 32 + ((l >> 4) * 16);
    const float* src = Wl + (size_t)n * (2 * HDIM) + kb;   // contiguous in k
    #pragma unroll
    for (int e = 0; e < 16; ++e) {
        __bf16 hi, lo;
        split2(src[e], hi, lo);
        Bhi[(size_t)w * 16 + e] = __builtin_bit_cast(unsigned short, hi);
        Blo[(size_t)w * 16 + e] = __builtin_bit_cast(unsigned short, lo);
    }
}

// ---------------------------------------------------------------------------
// Stage 3: out[m][n] = h[m,:] @ Wl1.T[:,n] + P[batch[m]][n]
// Grid: (NTOT/128, 2); 256 threads = 8 waves. Workgroup tile: 128 rows x 128
// cols. Each wave: 16 rows x 128 cols (8 accumulator tiles of v8f).
// K = 256 split into 8 chunks of 32; B staged via 64KB LDS in 2 phases.
// 3x bf16 WMMA per tile per chunk (hi*hi + lo*hi + hi*lo) ~= f32 precision.
// ---------------------------------------------------------------------------
__global__ __launch_bounds__(256) void gemm_kernel(
    const float* __restrict__ hmat, const int* __restrict__ batch,
    const float* __restrict__ P,
    const uint4* __restrict__ gBhi, const uint4* __restrict__ gBlo,
    float* __restrict__ out) {
    __shared__ uint4 ldsHi[2048];   // 32KB: 8 col-tiles x 4 k-chunks x 32 lanes x 32B
    __shared__ uint4 ldsLo[2048];   // 32KB

    const int tid    = threadIdx.x;
    const int lane   = tid & 31;
    const int wave   = tid >> 5;
    const int laneLo = lane & 15;
    const int laneHi = lane >> 4;
    const int hh      = blockIdx.y;            // column half (0 or 1)
    const int rowBase = blockIdx.x * 128 + wave * 16;
    const int colBase = hh * 128;

    v8f acc[8] = {};                           // 8 tiles of 16x16 f32

    const int rowA = rowBase + laneLo;         // A-operand row for this lane

    #pragma unroll
    for (int p = 0; p < 2; ++p) {
        if (p) __syncthreads();
        // Cooperative copy of this phase's B (hi+lo) into LDS.
        // LDS uint4 index == idx by construction of the global pack layout.
        #pragma unroll
        for (int i = 0; i < 8; ++i) {
            int idx = tid + i * 256;           // 0..2047
            int t   = idx >> 8;
            int cc  = (idx >> 6) & 3;
            int r   = idx & 63;
            int gidx = ((hh * 8 + t) * 8 + (p * 4 + cc)) * 64 + r;
            ldsHi[idx] = gBhi[gidx];
            ldsLo[idx] = gBlo[gidx];
        }
        __syncthreads();

        #pragma unroll
        for (int cc = 0; cc < 4; ++cc) {
            const int c  = p * 4 + cc;
            // A operand in this code's contiguous k order, same as B (16-bit A, 16x32): lane l = (half<<4)|m holds
            // row M = m, K = c*32 + half*16 + e.
            const int kb = c * 32 + laneHi * 16;
            const float* hp = hmat + (size_t)rowA * HDIM + kb;
            float4 x0 = *(const float4*)(hp +  0);
            float4 x1 = *(const float4*)(hp +  4);
            float4 x2 = *(const float4*)(hp +  8);
            float4 x3 = *(const float4*)(hp + 12);
            v16bf a_hi, a_lo;
            cvt4(x0,  0, a_hi, a_lo);
            cvt4(x1,  4, a_hi, a_lo);
            cvt4(x2,  8, a_hi, a_lo);
            cvt4(x3, 12, a_hi, a_lo);

            #pragma unroll
            for (int t = 0; t < 8; ++t) {
                const int base = ((t * 4 + cc) * 32 + lane) * 2;  // uint4 units
                union { uint4 q[2]; v16bf v; } bh, bl2;
                bh.q[0]  = ldsHi[base];  bh.q[1]  = ldsHi[base + 1];
                bl2.q[0] = ldsLo[base];  bl2.q[1] = ldsLo[base + 1];
                acc[t] = __builtin_amdgcn_wmma_f32_16x16x32_bf16(
                    false, a_hi, false, bh.v,  (short)0, acc[t], false, false);
                acc[t] = __builtin_amdgcn_wmma_f32_16x16x32_bf16(
                    false, a_lo, false, bh.v,  (short)0, acc[t], false, false);
                acc[t] = __builtin_amdgcn_wmma_f32_16x16x32_bf16(
                    false, a_hi, false, bl2.v, (short)0, acc[t], false, false);
            }
        }
    }

    // Epilogue: C/D layout — lane l, vgpr v => row M = v + 8*(l>>4), col N = l&15.
    int bidx[8];
    #pragma unroll
    for (int v = 0; v < 8; ++v)
        bidx[v] = batch[rowBase + laneHi * 8 + v];

    #pragma unroll
    for (int t = 0; t < 8; ++t) {
        const int col = colBase + t * 16 + laneLo;
        #pragma unroll
        for (int v = 0; v < 8; ++v) {
            const int row = rowBase + laneHi * 8 + v;
            out[(size_t)row * HDIM + col] =
                acc[t][v] + P[(size_t)bidx[v] * HDIM + col];
        }
    }
}

// ---------------------------------------------------------------------------
extern "C" void kernel_launch(void* const* d_in, const int* in_sizes, int n_in,
                              void* d_out, int out_size, void* d_ws, size_t ws_size,
                              hipStream_t stream) {
    (void)in_sizes; (void)n_in; (void)out_size; (void)ws_size;
    const float* hmat     = (const float*)d_in[0];   // [N, H]
    const float* ret_feat = (const float*)d_in[1];   // [B, K, R]
    const int*   batch    = (const int*)  d_in[2];   // [N]
    const float* Wp       = (const float*)d_in[3];   // [H, R]
    const float* bp       = (const float*)d_in[4];   // [H]
    const float* Wl       = (const float*)d_in[5];   // [H, 2H]
    const float* bl       = (const float*)d_in[6];   // [H]
    float* out = (float*)d_out;                      // [N, H]

    // Workspace: P (64KB) | Bhi (128KB) | Blo (128KB)  => 320KB total
    char* ws = (char*)d_ws;
    float*          P   = (float*)ws;
    unsigned short* Bhi = (unsigned short*)(ws + 64 * 1024);
    unsigned short* Blo = (unsigned short*)(ws + 64 * 1024 + 128 * 1024);

    pooled_kernel<<<BSZ, 256, 0, stream>>>(ret_feat, Wp, bp, Wl, bl, P);
    pack_kernel  <<<16,  256, 0, stream>>>(Wl, Bhi, Blo);

    dim3 grid(NTOT / 128, 2);
    gemm_kernel<<<grid, 256, 0, stream>>>(hmat, batch, P,
                                          (const uint4*)Bhi, (const uint4*)Blo, out);
}
